// TemporalSlotAttention_47407849013796
// MI455X (gfx1250) — matrix operand order repaired and checked by reading, not compiled, not run on hardware
//
#include <hip/hip_runtime.h>

#define DEV __device__ __forceinline__

typedef __attribute__((ext_vector_type(16))) __bf16 bf16x16;
typedef __attribute__((ext_vector_type(8)))  __bf16 bf16x8;
typedef __attribute__((ext_vector_type(8)))  float  f32x8;
typedef unsigned short u16;

// Problem sizes (fixed by the reference)
constexpr int CB = 16, CT = 8, CN = 1024, CF = 1024, CD = 512, CNS = 8, CHID = 512, CITERS = 3;
constexpr float CEPS   = 1e-8f;
constexpr float CSCALE = 0.04419417382415922f; // 512^-0.5

// ---------------------------------------------------------------------------
// 3x-bf16 emulated-f32 WMMA: (ah+al)*(bh+bl) ~= ah*bh + al*bh + ah*bl
// ---------------------------------------------------------------------------
DEV f32x8 wmma3(bf16x16 ah, bf16x16 al, bf16x16 bh, bf16x16 bl, f32x8 c) {
  c = __builtin_amdgcn_wmma_f32_16x16x32_bf16(false, ah, false, bh, (short)0, c, false, false);
  c = __builtin_amdgcn_wmma_f32_16x16x32_bf16(false, al, false, bh, (short)0, c, false, false);
  c = __builtin_amdgcn_wmma_f32_16x16x32_bf16(false, ah, false, bl, (short)0, c, false, false);
  return c;
}

// A-fragment (16x32 bf16, wave32): lane holds row (lane&15); K runs
// [half*16, half*16+16).  Source: LDS row-major,
// row stride 1024 elements.
DEV bf16x16 load_afrag_lds(const __bf16* s, int row, int k0, int half) {
  bf16x8 r0 = *(const bf16x8*)(s + row * 1024 + k0 + half * 16);
  bf16x8 r1 = *(const bf16x8*)(s + row * 1024 + k0 + half * 16 + 8);
  bf16x16 a;
#pragma unroll
  for (int i = 0; i < 8; ++i) { a[i] = r0[i]; a[8 + i] = r1[i]; }
  return a;
}

// ---------------------------------------------------------------------------
// Kernel A: per-frame K/V projection GEMM.
//   C[bn, c] = x[b,t,n,:] . Wkv[c,:] + bias,  c in [0,1024): c<512 -> k, else v
//   Output stored as bf16 hi/lo: k as [b][n][d], v transposed as [b][d][n].
//   Block tile 128x128, 8 waves (4Mx2N), wave tile 32x64 (2x4 WMMA frags).
// ---------------------------------------------------------------------------
__global__ __launch_bounds__(256) void gemm_kv(
    const float* __restrict__ x,
    const u16* __restrict__ wh_, const u16* __restrict__ wl_,
    const float* __restrict__ bk, const float* __restrict__ bv,
    u16* __restrict__ khi_, u16* __restrict__ klo_,
    u16* __restrict__ vthi_, u16* __restrict__ vtlo_, int t) {
  const __bf16* wh = (const __bf16*)wh_;
  const __bf16* wl = (const __bf16*)wl_;
  __bf16* khi = (__bf16*)khi_;  __bf16* klo = (__bf16*)klo_;
  __bf16* vthi = (__bf16*)vthi_; __bf16* vtlo = (__bf16*)vtlo_;

  const int lane = threadIdx.x & 31, wave = threadIdx.x >> 5;
  const int half = lane >> 4, lrow = lane & 15;
  const int waveM = wave >> 1, waveN = wave & 1;
  const int mbase = blockIdx.x * 128 + waveM * 32;   // global row (b*1024+n)
  const int nbase = blockIdx.y * 128 + waveN * 64;   // global col (k/v channel)

  f32x8 acc[2][4] = {};
  for (int k0 = 0; k0 < CF; k0 += 32) {
    bf16x16 ah[2], al[2];
#pragma unroll
    for (int ms = 0; ms < 2; ++ms) {
      const int rg = mbase + ms * 16 + lrow;
      const int bb = rg >> 10, nn = rg & 1023;
      const float* xr = x + (((size_t)bb * CT + t) * CN + nn) * CF + k0 + half * 16;
      float4 q0 = *(const float4*)(xr);
      float4 q1 = *(const float4*)(xr + 4);
      float4 q2 = *(const float4*)(xr + 8);
      float4 q3 = *(const float4*)(xr + 12);
      float fl[16] = {q0.x, q0.y, q0.z, q0.w, q1.x, q1.y, q1.z, q1.w,
                      q2.x, q2.y, q2.z, q2.w, q3.x, q3.y, q3.z, q3.w};
#pragma unroll
      for (int e = 0; e < 16; ++e) {
        __bf16 h = (__bf16)fl[e];
        ah[ms][e] = h;
        al[ms][e] = (__bf16)(fl[e] - (float)h);
      }
    }
#pragma unroll
    for (int ns = 0; ns < 4; ++ns) {
      const int col = nbase + ns * 16 + lrow;
      bf16x16 bh = *(const bf16x16*)(wh + (size_t)col * CF + k0 + half * 16);
      bf16x16 bl = *(const bf16x16*)(wl + (size_t)col * CF + k0 + half * 16);
#pragma unroll
      for (int ms = 0; ms < 2; ++ms)
        acc[ms][ns] = wmma3(ah[ms], al[ms], bh, bl, acc[ms][ns]);
    }
  }
  // Epilogue: C frag row = r + 8*half, col = lane&15; add bias, split, store.
#pragma unroll
  for (int ms = 0; ms < 2; ++ms) {
#pragma unroll
    for (int ns = 0; ns < 4; ++ns) {
      const int col = nbase + ns * 16 + lrow;
      const float bia = (col < CD) ? bk[col] : bv[col - CD];
#pragma unroll
      for (int r = 0; r < 8; ++r) {
        const int rg = mbase + ms * 16 + r + 8 * half;
        const int bb = rg >> 10, nn = rg & 1023;
        const float v = acc[ms][ns][r] + bia;
        __bf16 h = (__bf16)v;
        __bf16 l = (__bf16)(v - (float)h);
        if (col < CD) {
          const size_t o = ((size_t)bb * CN + nn) * CD + col;
          khi[o] = h; klo[o] = l;
        } else {
          const size_t o = ((size_t)bb * CD + (col - CD)) * CN + nn;
          vthi[o] = h; vtlo[o] = l;
        }
      }
    }
  }
}

// ---------------------------------------------------------------------------
// Kernel B helpers: stage f32 [8][C] -> bf16 hi/lo A buffer [16][1024]
// (rows 8..15 zeroed so M=8 rides in an M=16 WMMA tile).
// ---------------------------------------------------------------------------
DEV void convertA(__bf16* s_ahi, __bf16* s_alo, const float* src, int C,
                  const float* rinv, bool relu, float* copy_to) {
  for (int idx = threadIdx.x; idx < 16 * C; idx += (int)blockDim.x) {
    const int r = idx / C, c = idx - r * C;
    float v = 0.f;
    if (r < 8) {
      v = src[r * C + c];
      if (rinv) v *= rinv[r];
      if (relu) v = fmaxf(v, 0.f);
      if (copy_to) copy_to[r * C + c] = v;
    }
    __bf16 h = (__bf16)v;
    s_ahi[r * 1024 + c] = h;
    s_alo[r * 1024 + c] = (__bf16)(v - (float)h);
  }
}

// M=16 (8 live rows) x Nout GEMM: A from LDS staging, B from global bf16
// [Nout][K] row-major (columns of B contiguous in K).  16 waves, each owns
// TILES 16-wide output tiles.  Nout = 16*16*TILES.
template <int K, int TILES>
DEV void wave_mm(const __bf16* s_ahi, const __bf16* s_alo,
                 const __bf16* gbh, const __bf16* gbl,
                 const float* bias, float scale, const float* resid,
                 float* dst, int dstride) {
  const int lane = threadIdx.x & 31;
  const int wave = threadIdx.x >> 5;
  const int half = lane >> 4;
  const int lrow = lane & 15;
  f32x8 acc[TILES] = {};
  for (int k0 = 0; k0 < K; k0 += 32) {
    bf16x16 ah = load_afrag_lds(s_ahi, lrow, k0, half);
    bf16x16 al = load_afrag_lds(s_alo, lrow, k0, half);
#pragma unroll
    for (int tt = 0; tt < TILES; ++tt) {
      const int col = (wave * TILES + tt) * 16 + lrow;
      bf16x16 bh = *(const bf16x16*)(gbh + (size_t)col * K + k0 + half * 16);
      bf16x16 bl = *(const bf16x16*)(gbl + (size_t)col * K + k0 + half * 16);
      acc[tt] = wmma3(ah, al, bh, bl, acc[tt]);
    }
  }
  if (half == 0) {  // rows 0..7 live in lanes 0..15 (r + 8*half)
#pragma unroll
    for (int tt = 0; tt < TILES; ++tt) {
      const int col = (wave * TILES + tt) * 16 + lrow;
      const float bia = bias ? bias[col] : 0.f;
#pragma unroll
      for (int r = 0; r < 8; ++r) {
        float v = acc[tt][r] * scale + bia;
        if (resid) v += resid[r * dstride + col];
        dst[r * dstride + col] = v;
      }
    }
  }
}

// ---------------------------------------------------------------------------
// Kernel B: one workgroup per batch, 512 threads (16 wave32).  Runs the 3
// slot-attention iterations for frame t entirely on-chip.
// ---------------------------------------------------------------------------
__global__ __launch_bounds__(512) void slot_step(
    const u16* __restrict__ khi_, const u16* __restrict__ klo_,
    const u16* __restrict__ vthi_, const u16* __restrict__ vtlo_,
    const u16* __restrict__ wqh_, const u16* __restrict__ wql_,
    const u16* __restrict__ w1h_, const u16* __restrict__ w1l_,
    const u16* __restrict__ w2h_, const u16* __restrict__ w2l_,
    const float* __restrict__ bq, const float* __restrict__ b1,
    const float* __restrict__ b2,
    float* __restrict__ slots_g, float* __restrict__ out, int t) {
  __shared__ __bf16 s_ahi[16 * 1024];   // 32 KB  A-staging hi
  __shared__ __bf16 s_alo[16 * 1024];   // 32 KB  A-staging lo
  __shared__ float  s_f32[8 * 1024];    // 32 KB  stage output / dots / attn
  __shared__ float  s_slots[8 * 512];   // 16 KB  current slots (f32)
  __shared__ float  s_part[16 * 8];     // per-wave partial row sums
  __shared__ float  s_rinv[8];

  const int b = blockIdx.x;
  const int tid = threadIdx.x;
  const int lane = tid & 31, wave = tid >> 5;

  const __bf16* kh = (const __bf16*)khi_  + (size_t)b * CN * CD;
  const __bf16* kl = (const __bf16*)klo_  + (size_t)b * CN * CD;
  const __bf16* vh = (const __bf16*)vthi_ + (size_t)b * CD * CN;
  const __bf16* vl = (const __bf16*)vtlo_ + (size_t)b * CD * CN;
  const __bf16* wqh = (const __bf16*)wqh_; const __bf16* wql = (const __bf16*)wql_;
  const __bf16* w1h = (const __bf16*)w1h_; const __bf16* w1l = (const __bf16*)w1l_;
  const __bf16* w2h = (const __bf16*)w2h_; const __bf16* w2l = (const __bf16*)w2l_;

  for (int idx = tid; idx < CNS * CD; idx += 512)
    s_slots[idx] = slots_g[(size_t)b * CNS * CD + idx];
  __syncthreads();

  for (int it = 0; it < CITERS; ++it) {
    // q = slots @ Wq^T + bq
    convertA(s_ahi, s_alo, s_slots, CD, nullptr, false, nullptr);
    __syncthreads();
    wave_mm<CD, 2>(s_ahi, s_alo, wqh, wql, bq, 1.f, nullptr, s_f32, CD);
    __syncthreads();
    convertA(s_ahi, s_alo, s_f32, CD, nullptr, false, nullptr);
    __syncthreads();
    // dots = q @ k_t^T * scale   (Nout = N = 1024, K = D = 512)
    wave_mm<CD, 4>(s_ahi, s_alo, kh, kl, nullptr, CSCALE, nullptr, s_f32, CN);
    __syncthreads();
    // competitive softmax over slot axis, then renorm over keys
    {
      float part[8];
#pragma unroll
      for (int i = 0; i < 8; ++i) part[i] = 0.f;
      for (int j = tid; j < CN; j += 512) {
        float vals[8], m = -3.0e38f;
#pragma unroll
        for (int i = 0; i < 8; ++i) { vals[i] = s_f32[i * CN + j]; m = fmaxf(m, vals[i]); }
        float sum = 0.f;
#pragma unroll
        for (int i = 0; i < 8; ++i) { vals[i] = __expf(vals[i] - m); sum += vals[i]; }
        const float inv = 1.f / sum;
#pragma unroll
        for (int i = 0; i < 8; ++i) {
          const float a = vals[i] * inv + CEPS;
          s_f32[i * CN + j] = a;
          part[i] += a;
        }
      }
#pragma unroll
      for (int off = 16; off > 0; off >>= 1)
#pragma unroll
        for (int i = 0; i < 8; ++i) part[i] += __shfl_xor(part[i], off, 32);
      if (lane == 0)
#pragma unroll
        for (int i = 0; i < 8; ++i) s_part[wave * 8 + i] = part[i];
      __syncthreads();
      if (tid < 8) {
        float s = 0.f;
#pragma unroll
        for (int w = 0; w < 16; ++w) s += s_part[w * 8 + tid];
        s_rinv[tid] = 1.f / s;
      }
      __syncthreads();
    }
    // attn (normalized) -> staging; slots_upd = attn @ v_t  (K = N = 1024)
    convertA(s_ahi, s_alo, s_f32, CN, s_rinv, false, nullptr);
    __syncthreads();
    wave_mm<CN, 2>(s_ahi, s_alo, vh, vl, nullptr, 1.f, nullptr, s_f32, CD);
    __syncthreads();
    convertA(s_ahi, s_alo, s_f32, CD, nullptr, false, s_slots);  // keep residual
    __syncthreads();
    // hdn = relu(slots_upd @ W1^T + b1)
    wave_mm<CD, 2>(s_ahi, s_alo, w1h, w1l, b1, 1.f, nullptr, s_f32, CHID);
    __syncthreads();
    convertA(s_ahi, s_alo, s_f32, CHID, nullptr, true, nullptr);
    __syncthreads();
    // slots = hdn @ W2^T + b2 + slots_upd
    wave_mm<CHID, 2>(s_ahi, s_alo, w2h, w2l, b2, 1.f, s_slots, s_slots, CD);
    __syncthreads();
  }

  for (int idx = tid; idx < CNS * CD; idx += 512) {
    const float v = s_slots[idx];
    slots_g[(size_t)b * CNS * CD + idx] = v;
    const int i = idx >> 9, d = idx & 511;
    out[(((size_t)b * CT + t) * CNS + i) * CD + d] = v;
  }
}

// ---------------------------------------------------------------------------
// Prep kernels
// ---------------------------------------------------------------------------
__global__ void split_bf16(const float* __restrict__ src, u16* __restrict__ hi_,
                           u16* __restrict__ lo_, int n) {
  __bf16* hi = (__bf16*)hi_; __bf16* lo = (__bf16*)lo_;
  const int i = blockIdx.x * blockDim.x + threadIdx.x;
  if (i < n) {
    const float x = src[i];
    __bf16 h = (__bf16)x;
    hi[i] = h;
    lo[i] = (__bf16)(x - (float)h);
  }
}

__global__ void copy_f32(const float* __restrict__ src, float* __restrict__ dst, int n) {
  const int i = blockIdx.x * blockDim.x + threadIdx.x;
  if (i < n) dst[i] = src[i];
}

// ---------------------------------------------------------------------------
// Launcher
// ---------------------------------------------------------------------------
extern "C" void kernel_launch(void* const* d_in, const int* in_sizes, int n_in,
                              void* d_out, int out_size, void* d_ws, size_t ws_size,
                              hipStream_t stream) {
  (void)in_sizes; (void)n_in; (void)out_size; (void)ws_size;
  const float* x   = (const float*)d_in[0];
  const float* ini = (const float*)d_in[1];
  const float* Wq  = (const float*)d_in[2];
  const float* bq  = (const float*)d_in[3];
  const float* Wk  = (const float*)d_in[4];
  const float* bk  = (const float*)d_in[5];
  const float* Wv  = (const float*)d_in[6];
  const float* bv  = (const float*)d_in[7];
  const float* W1  = (const float*)d_in[8];
  const float* b1  = (const float*)d_in[9];
  const float* W2  = (const float*)d_in[10];
  const float* b2  = (const float*)d_in[11];
  float* out = (float*)d_out;

  char* ws = (char*)d_ws;
  auto alloc = [&](size_t bytes) -> char* {
    char* p = ws;
    ws += (bytes + 255) & ~(size_t)255;
    return p;
  };
  // Weight splits (bf16 hi/lo), [out][reduction] row-major
  u16* wkvh = (u16*)alloc((size_t)2 * CD * CF * 2);  // Wk rows then Wv rows
  u16* wkvl = (u16*)alloc((size_t)2 * CD * CF * 2);
  u16* wqh  = (u16*)alloc((size_t)CD * CD * 2);
  u16* wql  = (u16*)alloc((size_t)CD * CD * 2);
  u16* w1h  = (u16*)alloc((size_t)CHID * CD * 2);
  u16* w1l  = (u16*)alloc((size_t)CHID * CD * 2);
  u16* w2h  = (u16*)alloc((size_t)CD * CHID * 2);
  u16* w2l  = (u16*)alloc((size_t)CD * CHID * 2);
  // Per-frame K/V (bf16 hi/lo): k[b][n][d], vT[b][d][n]
  u16* khi  = (u16*)alloc((size_t)CB * CN * CD * 2);
  u16* klo  = (u16*)alloc((size_t)CB * CN * CD * 2);
  u16* vthi = (u16*)alloc((size_t)CB * CD * CN * 2);
  u16* vtlo = (u16*)alloc((size_t)CB * CD * CN * 2);
  // Persistent slots f32 [B][NS][D]
  float* slots_g = (float*)alloc((size_t)CB * CNS * CD * 4);

  const int TPB = 256;
  auto blocks = [&](int n) { return (n + TPB - 1) / TPB; };

  const int nW = CD * CF;  // 512*1024
  split_bf16<<<blocks(nW), TPB, 0, stream>>>(Wk, wkvh, wkvl, nW);
  split_bf16<<<blocks(nW), TPB, 0, stream>>>(Wv, wkvh + (size_t)nW, wkvl + (size_t)nW, nW);
  split_bf16<<<blocks(CD * CD), TPB, 0, stream>>>(Wq, wqh, wql, CD * CD);
  split_bf16<<<blocks(CHID * CD), TPB, 0, stream>>>(W1, w1h, w1l, CHID * CD);
  split_bf16<<<blocks(CD * CHID), TPB, 0, stream>>>(W2, w2h, w2l, CD * CHID);
  copy_f32<<<blocks(CB * CNS * CD), TPB, 0, stream>>>(ini, slots_g, CB * CNS * CD);

  for (int t = 0; t < CT; ++t) {
    gemm_kv<<<dim3((CB * CN) / 128, (2 * CD) / 128), 256, 0, stream>>>(
        x, wkvh, wkvl, bk, bv, khi, klo, vthi, vtlo, t);
    slot_step<<<CB, 512, 0, stream>>>(khi, klo, vthi, vtlo, wqh, wql, w1h, w1l,
                                      w2h, w2l, bq, b1, b2, slots_g, out, t);
  }
}
